// SOLOV2_decode_53489522704498
// MI455X (gfx1250) — 8-bit matrix operand order repaired and checked by reading, compiled, not run on hardware
//
#include <hip/hip_runtime.h>
#include <hip/hip_bf16.h>
#include <math.h>

typedef int v8i __attribute__((ext_vector_type(8)));

#define N_CELLS   3872
#define NUM_CLS   80
#define NC        (N_CELLS * NUM_CLS)
#define NPAD      524288
#define FH        128
#define FW        192
#define SEG_HW    (FH * FW)
#define NMS_PRE   500
#define MPAD      512
#define MAXIMG    100
#define IMG_H     512
#define IMG_W     768
#define OUT_HW    (IMG_H * IMG_W)

__device__ __forceinline__ unsigned int enc_f32(float f) {
  unsigned int b = __float_as_uint(f);
  return (b & 0x80000000u) ? ~b : (b | 0x80000000u);
}
__device__ __forceinline__ float dec_f32(unsigned int e) {
  unsigned int b = (e & 0x80000000u) ? (e & 0x7FFFFFFFu) : ~e;
  return __uint_as_float(b);
}

__global__ __launch_bounds__(256) void k_init_keys(const float* __restrict__ cate,
                                                   unsigned long long* __restrict__ keys) {
  int i = blockIdx.x * blockDim.x + threadIdx.x;
  if (i >= NPAD) return;
  unsigned long long k = 0ull;
  if (i < NC) {
    float f = cate[i];
    float cand = (f > 0.1f) ? f : -1.0f;
    k = ((unsigned long long)enc_f32(cand) << 32) | (unsigned long long)(0xFFFFFFFFu - (unsigned)i);
  }
  keys[i] = k;
}

__global__ __launch_bounds__(256) void k_bitonic(unsigned long long* __restrict__ keys, int j, int k) {
  int i = blockIdx.x * blockDim.x + threadIdx.x;
  if (i >= NPAD) return;
  int ixj = i ^ j;
  if (ixj > i) {
    unsigned long long a = keys[i], b = keys[ixj];
    bool desc = ((i & k) == 0);
    bool sw = desc ? (a < b) : (a > b);
    if (sw) { keys[i] = b; keys[ixj] = a; }
  }
}

__global__ __launch_bounds__(256) void k_cand_eval(const unsigned long long* __restrict__ keys,
                                                   const float* __restrict__ seg,
                                                   float* __restrict__ cand_score,
                                                   int* __restrict__ cand_label,
                                                   int* __restrict__ cand_grid,
                                                   float* __restrict__ cand_sum) {
  __shared__ float s_cnt[256];
  __shared__ float s_sum[256];
  int i = blockIdx.x;
  int t = threadIdx.x;
  unsigned long long key = keys[i];
  float sc = dec_f32((unsigned int)(key >> 32));
  unsigned idx = 0xFFFFFFFFu - (unsigned)(key & 0xFFFFFFFFu);
  int grid = (int)(idx / NUM_CLS);
  int lab  = (int)(idx % NUM_CLS);
  const float* S = seg + (size_t)grid * SEG_HW;
  float cnt = 0.f, ssum = 0.f;
  for (int p = t; p < SEG_HW; p += 256) {
    float v = S[p];
    if (v > 0.5f) { cnt += 1.f; ssum += v; }
  }
  s_cnt[t] = cnt; s_sum[t] = ssum;
  __syncthreads();
  for (int s = 128; s > 0; s >>= 1) {
    if (t < s) { s_cnt[t] += s_cnt[t + s]; s_sum[t] += s_sum[t + s]; }
    __syncthreads();
  }
  if (t == 0) {
    float count = s_cnt[0];
    float strd = (grid < 2896) ? 8.f : ((grid < 3472) ? 16.f : 32.f);
    bool valid = sc > 0.1f;
    bool keep  = valid && (count > strd);
    float segscore = s_sum[0] / fmaxf(count, 1.f);
    cand_score[i] = keep ? sc * segscore : 0.f;
    cand_label[i] = lab;
    cand_grid[i]  = grid;
    cand_sum[i]   = count;
  }
}

__device__ void sort512_lds(unsigned long long* k, int t) {
  for (int size = 2; size <= 512; size <<= 1) {
    for (int stride = size >> 1; stride > 0; stride >>= 1) {
      int j = t ^ stride;
      if (j > t) {
        unsigned long long a = k[t], b = k[j];
        bool desc = ((t & size) == 0);
        bool sw = desc ? (a < b) : (a > b);
        if (sw) { k[t] = b; k[j] = a; }
      }
      __syncthreads();
    }
  }
}

__global__ __launch_bounds__(512) void k_sort1_gather(const float* __restrict__ cand_score,
                                                      const int* __restrict__ cand_label,
                                                      const int* __restrict__ cand_grid,
                                                      const float* __restrict__ cand_sum,
                                                      int* __restrict__ perm1,
                                                      float* __restrict__ s_score,
                                                      int* __restrict__ s_label,
                                                      int* __restrict__ s_grid,
                                                      float* __restrict__ s_sum) {
  __shared__ unsigned long long k[512];
  int t = threadIdx.x;
  unsigned int e = (t < NMS_PRE) ? enc_f32(cand_score[t]) : 0u;
  k[t] = ((unsigned long long)e << 32) | (unsigned long long)(0xFFFFFFFFu - (unsigned)t);
  __syncthreads();
  sort512_lds(k, t);
  int p = (int)(0xFFFFFFFFu - (unsigned)(k[t] & 0xFFFFFFFFu));
  perm1[t]  = p;
  s_score[t] = cand_score[p];
  s_label[t] = cand_label[p];
  s_grid[t]  = cand_grid[p];
  s_sum[t]   = cand_sum[p];
}

__global__ __launch_bounds__(256) void k_build_mask(const float* __restrict__ seg,
                                                    const int* __restrict__ s_grid,
                                                    unsigned char* __restrict__ M) {
  int r = blockIdx.x;
  int t = threadIdx.x;
  unsigned char* row = M + (size_t)r * SEG_HW;
  if (r < NMS_PRE) {
    const float* S = seg + (size_t)s_grid[r] * SEG_HW;
    for (int p = t; p < SEG_HW; p += 256) row[p] = (S[p] > 0.5f) ? 1 : 0;
  } else {
    for (int p = t; p < SEG_HW; p += 256) row[p] = 0;
  }
}

__global__ __launch_bounds__(32) void k_nms_gemm(const unsigned char* __restrict__ M,
                                                 int* __restrict__ inter) {
  const int K = SEG_HW;
  int tile = blockIdx.x;
  int tr = tile >> 5;
  int tc = tile & 31;
  int lane = threadIdx.x;
  int mrow = lane & 15;
  int hi = (lane >> 4) & 1;
  const unsigned char* Arow = M + (size_t)(tr * 16 + mrow) * K;
  const unsigned char* Brow = M + (size_t)(tc * 16 + mrow) * K;
  v8i acc = {};
  for (int kk = 0; kk < K; kk += 64) {
    int ab = kk + hi * 16;
    uint2 p0 = *(const uint2*)(Arow + ab);
    uint2 p1 = *(const uint2*)(Arow + ab + 8);
    uint2 p2 = *(const uint2*)(Arow + ab + 32);
    uint2 p3 = *(const uint2*)(Arow + ab + 40);
    int bb = kk + hi * 16;
    uint4 q0 = *(const uint4*)(Brow + bb);
    uint4 q1 = *(const uint4*)(Brow + bb + 32);
    __builtin_prefetch(Arow + ab + 64, 0, 1);
    v8i a, b;
    a[0] = (int)p0.x; a[1] = (int)p0.y; a[2] = (int)p1.x; a[3] = (int)p1.y;
    a[4] = (int)p2.x; a[5] = (int)p2.y; a[6] = (int)p3.x; a[7] = (int)p3.y;
    b[0] = (int)q0.x; b[1] = (int)q0.y; b[2] = (int)q0.z; b[3] = (int)q0.w;
    b[4] = (int)q1.x; b[5] = (int)q1.y; b[6] = (int)q1.z; b[7] = (int)q1.w;
    acc = __builtin_amdgcn_wmma_i32_16x16x64_iu8(false, a, false, b, acc, false, false);
  }
  int col = tc * 16 + (lane & 15);
  int rbase = tr * 16 + hi * 8;
  for (int r = 0; r < 8; ++r) inter[(size_t)(rbase + r) * MPAD + col] = acc[r];
}

__global__ __launch_bounds__(256) void k_comp(const int* __restrict__ inter,
                                              const int* __restrict__ s_label,
                                              const float* __restrict__ s_sum,
                                              float* __restrict__ comp) {
  __shared__ float red[256];
  int j = blockIdx.x;
  int t = threadIdx.x;
  int labj = s_label[j];
  float sumj = s_sum[j];
  float mx = 0.f;
  for (int i = t; i < NMS_PRE; i += 256) {
    if (i < j && s_label[i] == labj) {
      float it = (float)inter[(size_t)i * MPAD + j];
      float un = s_sum[i] + sumj - it;
      float iou = it / fmaxf(un, 1e-6f);
      mx = fmaxf(mx, iou);
    }
  }
  red[t] = mx;
  __syncthreads();
  for (int s = 128; s > 0; s >>= 1) {
    if (t < s) red[t] = fmaxf(red[t], red[t + s]);
    __syncthreads();
  }
  if (t == 0) comp[j] = red[0];
}

__global__ __launch_bounds__(256) void k_coef(const int* __restrict__ inter,
                                              const int* __restrict__ s_label,
                                              const float* __restrict__ s_sum,
                                              const float* __restrict__ comp,
                                              const float* __restrict__ s_score,
                                              float* __restrict__ scores2) {
  __shared__ float red[256];
  int j = blockIdx.x;
  int t = threadIdx.x;
  if (j >= NMS_PRE) { if (t == 0) scores2[j] = 0.f; return; }
  int labj = s_label[j];
  float sumj = s_sum[j];
  float mn = 3.4e38f;
  for (int i = t; i < NMS_PRE; i += 256) {
    float d = 0.f;
    if (i < j && s_label[i] == labj) {
      float it = (float)inter[(size_t)i * MPAD + j];
      float un = s_sum[i] + sumj - it;
      d = it / fmaxf(un, 1e-6f);
    }
    float ci = comp[i];
    float v = expf(-(d * d - ci * ci) * 0.5f);
    mn = fminf(mn, v);
  }
  red[t] = mn;
  __syncthreads();
  for (int s = 128; s > 0; s >>= 1) {
    if (t < s) red[t] = fminf(red[t], red[t + s]);
    __syncthreads();
  }
  if (t == 0) {
    float ns = s_score[j] * red[0];
    scores2[j] = (ns >= 0.05f) ? ns : 0.f;
  }
}

__global__ __launch_bounds__(512) void k_sort2(const float* __restrict__ scores2,
                                               int* __restrict__ perm2) {
  __shared__ unsigned long long k[512];
  int t = threadIdx.x;
  unsigned int e = (t < NMS_PRE) ? enc_f32(scores2[t]) : 0u;
  k[t] = ((unsigned long long)e << 32) | (unsigned long long)(0xFFFFFFFFu - (unsigned)t);
  __syncthreads();
  sort512_lds(k, t);
  perm2[t] = (int)(0xFFFFFFFFu - (unsigned)(k[t] & 0xFFFFFFFFu));
}

__global__ __launch_bounds__(128) void k_gather_out(const float* __restrict__ scores2,
                                                    const int* __restrict__ s_label,
                                                    const int* __restrict__ s_grid,
                                                    const int* __restrict__ perm2,
                                                    float* __restrict__ out,
                                                    int* __restrict__ out_grid) {
  int t = threadIdx.x;
  if (t >= MAXIMG) return;
  int slot = perm2[t];
  out[t]           = scores2[slot];
  out[MAXIMG + t]  = (float)s_label[slot];
  out_grid[t]      = s_grid[slot];
}

__global__ __launch_bounds__(256) void k_upsample(const float* __restrict__ seg,
                                                  const int* __restrict__ out_grid,
                                                  float* __restrict__ outm) {
  long long idx = (long long)blockIdx.x * blockDim.x + threadIdx.x;
  const long long total = (long long)MAXIMG * OUT_HW;
  if (idx >= total) return;
  int n = (int)(idx / OUT_HW);
  int rem = (int)(idx % OUT_HW);
  int y = rem / IMG_W, x = rem % IMG_W;
  const float* S = seg + (size_t)out_grid[n] * SEG_HW;
  float sy = fminf(fmaxf((y + 0.5f) * 0.25f - 0.5f, 0.f), (float)(FH - 1));
  float sx = fminf(fmaxf((x + 0.5f) * 0.25f - 0.5f, 0.f), (float)(FW - 1));
  int y0 = (int)sy; int y1 = min(y0 + 1, FH - 1); float fy = sy - (float)y0;
  int x0 = (int)sx; int x1 = min(x0 + 1, FW - 1); float fx = sx - (float)x0;
  float v00 = S[y0 * FW + x0], v01 = S[y0 * FW + x1];
  float v10 = S[y1 * FW + x0], v11 = S[y1 * FW + x1];
  float v = v00 * (1.f - fy) * (1.f - fx) + v01 * (1.f - fy) * fx
          + v10 * fy * (1.f - fx) + v11 * fy * fx;
  outm[idx] = (v > 0.5f) ? 1.f : 0.f;
}

__global__ __launch_bounds__(256) void k_boxes(const float* __restrict__ masks,
                                               const float* __restrict__ scores,
                                               float* __restrict__ boxes) {
  __shared__ int r_xmin[256], r_xmax[256], r_ymin[256], r_ymax[256];
  int n = blockIdx.x;
  int t = threadIdx.x;
  const float* Mn = masks + (size_t)n * OUT_HW;
  int xmin = IMG_W, xmax = 0, ymin = IMG_H, ymax = 0;
  for (int p = t; p < OUT_HW; p += 256) {
    if (Mn[p] > 0.5f) {
      int y = p / IMG_W, x = p % IMG_W;
      xmin = min(xmin, x); xmax = max(xmax, x);
      ymin = min(ymin, y); ymax = max(ymax, y);
    }
  }
  r_xmin[t] = xmin; r_xmax[t] = xmax; r_ymin[t] = ymin; r_ymax[t] = ymax;
  __syncthreads();
  for (int s = 128; s > 0; s >>= 1) {
    if (t < s) {
      r_xmin[t] = min(r_xmin[t], r_xmin[t + s]);
      r_xmax[t] = max(r_xmax[t], r_xmax[t + s]);
      r_ymin[t] = min(r_ymin[t], r_ymin[t + s]);
      r_ymax[t] = max(r_ymax[t], r_ymax[t + s]);
    }
    __syncthreads();
  }
  if (t == 0) {
    float g = (scores[n] > 0.f) ? 1.f : 0.f;
    boxes[n * 4 + 0] = (float)r_xmin[0] * g;
    boxes[n * 4 + 1] = (float)r_ymin[0] * g;
    boxes[n * 4 + 2] = (float)r_xmax[0] * g;
    boxes[n * 4 + 3] = (float)r_ymax[0] * g;
  }
}

extern "C" void kernel_launch(void* const* d_in, const int* in_sizes, int n_in,
                              void* d_out, int out_size, void* d_ws, size_t ws_size,
                              hipStream_t stream) {
  const float* cate = (const float*)d_in[0];
  const float* seg  = (const float*)d_in[1];
  float* out = (float*)d_out;

  char* ws = (char*)d_ws;
  size_t off = 0;
  auto carve = [&](size_t bytes) { char* p = ws + off; off += (bytes + 255) & ~(size_t)255; return p; };
  unsigned long long* keys = (unsigned long long*)carve((size_t)NPAD * 8);
  float* cand_score = (float*)carve(MPAD * 4);
  int*   cand_label = (int*)  carve(MPAD * 4);
  int*   cand_grid  = (int*)  carve(MPAD * 4);
  float* cand_sum   = (float*)carve(MPAD * 4);
  int*   perm1      = (int*)  carve(MPAD * 4);
  float* s_score    = (float*)carve(MPAD * 4);
  int*   s_label    = (int*)  carve(MPAD * 4);
  int*   s_grid     = (int*)  carve(MPAD * 4);
  float* s_sum      = (float*)carve(MPAD * 4);
  float* comp       = (float*)carve(MPAD * 4);
  float* scores2    = (float*)carve(MPAD * 4);
  int*   perm2      = (int*)  carve(MPAD * 4);
  int*   out_grid   = (int*)  carve(128 * 4);
  unsigned char* Mu8 = (unsigned char*)carve((size_t)MPAD * SEG_HW);
  int*   inter      = (int*)  carve((size_t)MPAD * MPAD * 4);
  (void)ws_size; (void)in_sizes; (void)n_in; (void)out_size;

  k_init_keys<<<NPAD / 256, 256, 0, stream>>>(cate, keys);
  for (int k = 2; k <= NPAD; k <<= 1)
    for (int j = k >> 1; j > 0; j >>= 1)
      k_bitonic<<<NPAD / 256, 256, 0, stream>>>(keys, j, k);

  k_cand_eval<<<MPAD, 256, 0, stream>>>(keys, seg, cand_score, cand_label, cand_grid, cand_sum);

  k_sort1_gather<<<1, 512, 0, stream>>>(cand_score, cand_label, cand_grid, cand_sum,
                                        perm1, s_score, s_label, s_grid, s_sum);

  k_build_mask<<<MPAD, 256, 0, stream>>>(seg, s_grid, Mu8);
  k_nms_gemm<<<(MPAD / 16) * (MPAD / 16), 32, 0, stream>>>(Mu8, inter);

  k_comp<<<NMS_PRE, 256, 0, stream>>>(inter, s_label, s_sum, comp);
  k_coef<<<MPAD, 256, 0, stream>>>(inter, s_label, s_sum, comp, s_score, scores2);

  k_sort2<<<1, 512, 0, stream>>>(scores2, perm2);
  k_gather_out<<<1, 128, 0, stream>>>(scores2, s_label, s_grid, perm2, out, out_grid);

  float* outm = out + 2 * MAXIMG;
  long long total = (long long)MAXIMG * OUT_HW;
  k_upsample<<<(int)((total + 255) / 256), 256, 0, stream>>>(seg, out_grid, outm);

  float* boxes = outm + total;
  k_boxes<<<MAXIMG, 256, 0, stream>>>(outm, out, boxes);
}
